// TVADE_block_44212393345559
// MI455X (gfx1250) — matrix operand order repaired and checked by reading, not compiled, not run on hardware
//
#include <hip/hip_runtime.h>
#include <hip/hip_bf16.h>
#include <math.h>

typedef _Float16 f16;
typedef __attribute__((ext_vector_type(16))) _Float16 v16h;
typedef __attribute__((ext_vector_type(8)))  _Float16 v8h;
typedef __attribute__((ext_vector_type(8)))  float    v8f;

#define GTM 64
#define GTN 256
#define GKT 32

// ---------------------------------------------------------------------------
// CDNA5 async copy: global -> LDS, 16 bytes per lane, tracked by ASYNCcnt.
// ---------------------------------------------------------------------------
__device__ __forceinline__ void async_copy_b128(const void* gptr, void* lptr)
{
  unsigned lds = (unsigned)(unsigned long long)lptr;   // flat->LDS: low 32 bits
  unsigned long long g = (unsigned long long)gptr;
  asm volatile("global_load_async_to_lds_b128 %0, %1, off"
               :: "v"(lds), "v"(g) : "memory");
}
__device__ __forceinline__ void wait_async0()
{
  asm volatile("s_wait_asynccnt 0" ::: "memory");
}

// ---------------------------------------------------------------------------
// Batched GEMM via V_WMMA_F32_16X16X32_F16, double-buffered LDS staging.
//   C[b] (MxN fp32, optional) / Ch[b] (f16 copy, optional)
//     = alpha * A[b](MxK) * op(B[b]) + bias[col]
//   TRANSB=1: Bm is [N,K] row-major (B^T), staged with async LDS copies.
//   TRANSB=0: Bm is [K,N] row-major, staged via scalar LDS transpose.
//   Requires M%64==0, N%256==0, K%32==0 (true for all shapes here).
//   Block: 256 threads = 8 waves (2 along M x 4 along N), wave tile 32x64.
// ---------------------------------------------------------------------------
template<int TRANSB>
__global__ __launch_bounds__(256) void k_gemm_wmma(
    const f16* __restrict__ A, int lda, long long sA,
    const f16* __restrict__ Bm, int ldb, long long sB,
    const float* __restrict__ bias, float alpha,
    float* __restrict__ C, f16* __restrict__ Ch, long long sC,
    int M, int N, int K)
{
  const int bz = blockIdx.z;
  A  += (long long)bz * sA;
  Bm += (long long)bz * sB;
  const int tid  = threadIdx.x;
  const int lane = tid & 31;
  const int wv   = tid >> 5;     // 0..7
  const int grp  = lane >> 4;    // 0/1 (lane half)
  const int idx  = lane & 15;
  const int waveM = wv >> 2;     // 2 waves along M
  const int waveN = wv & 3;      // 4 waves along N
  const int rowBase = blockIdx.y * GTM;
  const int colBase = blockIdx.x * GTN;

  __shared__ alignas(32) f16 As[2][GTM * GKT];   // [m][k]
  __shared__ alignas(32) f16 Bs[2][GTN * GKT];   // [n][k]

  v8f acc[2][4];
  for (int i = 0; i < 2; ++i)
    for (int j = 0; j < 4; ++j)
      for (int e = 0; e < 8; ++e) acc[i][j][e] = 0.0f;

  auto stage = [&](int k0, int buf) {
    { // A tile: 64 rows x 32 halves = 4 KB, one async b128 per thread
      int r = tid >> 2, c4 = tid & 3;
      async_copy_b128(A + (long long)(rowBase + r) * lda + k0 + c4 * 8,
                      As[buf] + r * GKT + c4 * 8);
    }
    if (TRANSB) { // B tile: 256 rows x 32 halves = 16 KB, 4 async b128/thread
      for (int i = 0; i < 4; ++i) {
        int lin = tid + i * 256;
        int r = lin >> 2, c4 = lin & 3;
        async_copy_b128(Bm + (long long)(colBase + r) * ldb + k0 + c4 * 8,
                        Bs[buf] + r * GKT + c4 * 8);
      }
    } else {      // B tile [32 k][256 n] -> LDS transpose to Bs[n][k]
      for (int i = 0; i < 32; ++i) {
        int lin = tid + i * 256;
        int kk = lin >> 8, nn = lin & 255;
        Bs[buf][nn * GKT + kk] = Bm[(long long)(k0 + kk) * ldb + colBase + nn];
      }
    }
  };

  stage(0, 0);
  const int ktiles = K / GKT;
  for (int t = 0; t < ktiles; ++t) {
    wait_async0();
    __syncthreads();
    if (t + 1 < ktiles) stage((t + 1) * GKT, (t + 1) & 1);

    const f16* as = As[t & 1];
    const f16* bs = Bs[t & 1];
    v16h af[2], bf[4];
    for (int tm = 0; tm < 2; ++tm) {
      // A 16x32 f16 layout: lane half g holds K = [g*8..g*8+7] ++ [16+g*8..16+g*8+7]
      const f16* rp = as + (waveM * 32 + tm * 16 + idx) * GKT;
      v8h lo = *(const v8h*)(rp + grp * 8);
      v8h hi = *(const v8h*)(rp + 16 + grp * 8);
      af[tm] = __builtin_shufflevector(lo, hi, 0,1,2,3,4,5,6,7,8,9,10,11,12,13,14,15);
    }
    for (int tn = 0; tn < 4; ++tn) {
      // B 32x16 f16 layout: lane half g holds K = [g*8..g*8+7] ++ [16+g*8..16+g*8+7]
      const f16* cp = bs + (waveN * 64 + tn * 16 + idx) * GKT;
      v8h blo = *(const v8h*)(cp + grp * 8);
      v8h bhi = *(const v8h*)(cp + 16 + grp * 8);
      bf[tn] = __builtin_shufflevector(blo, bhi, 0,1,2,3,4,5,6,7,8,9,10,11,12,13,14,15);
    }
    for (int tm = 0; tm < 2; ++tm)
      for (int tn = 0; tn < 4; ++tn)
        acc[tm][tn] = __builtin_amdgcn_wmma_f32_16x16x32_f16(
            false, af[tm], false, bf[tn], (short)0, acc[tm][tn], false, false);
  }

  // epilogue: D VGPR r -> row = r + 8*grp, col = idx
  const long long cb = (long long)bz * sC;
  for (int tm = 0; tm < 2; ++tm)
    for (int tn = 0; tn < 4; ++tn) {
      int col = colBase + waveN * 64 + tn * 16 + idx;
      float bvv = bias ? bias[col] : 0.0f;
      for (int r = 0; r < 8; ++r) {
        int row = rowBase + waveM * 32 + tm * 16 + grp * 8 + r;
        float v = acc[tm][tn][r] * alpha + bvv;
        long long o = cb + (long long)row * N + col;
        if (C)  C[o]  = v;
        if (Ch) Ch[o] = (f16)v;
      }
    }
}

// ---------------------------------------------------------------------------
// Row softmax: fp32 in -> f16 out, one 256-thread block per row.
// ---------------------------------------------------------------------------
__global__ __launch_bounds__(256) void k_softmax_rows(
    const float* __restrict__ in, f16* __restrict__ out, int ncol)
{
  long long row = blockIdx.x;
  const float* r = in + row * (long long)ncol;
  f16* w = out + row * (long long)ncol;
  __shared__ float sred[8];
  int tid = threadIdx.x, lane = tid & 31, wv = tid >> 5;

  float m = -3.4e38f;
  for (int c = tid; c < ncol; c += 256) m = fmaxf(m, r[c]);
  for (int off = 16; off >= 1; off >>= 1) m = fmaxf(m, __shfl_xor(m, off, 32));
  if (lane == 0) sred[wv] = m;
  __syncthreads();
  float bm = sred[0];
  for (int i = 1; i < 8; ++i) bm = fmaxf(bm, sred[i]);
  __syncthreads();

  float s = 0.0f;
  for (int c = tid; c < ncol; c += 256) s += __expf(r[c] - bm);
  for (int off = 16; off >= 1; off >>= 1) s += __shfl_xor(s, off, 32);
  if (lane == 0) sred[wv] = s;
  __syncthreads();
  float bs = 0.0f;
  for (int i = 0; i < 8; ++i) bs += sred[i];
  float inv = 1.0f / bs;
  for (int c = tid; c < ncol; c += 256) w[c] = (f16)(__expf(r[c] - bm) * inv);
}

// ---------------------------------------------------------------------------
// out = LayerNorm(x + res) * g + b over rows of length Lc; fp32 out + optional
// f16 copy. One block per row.
// ---------------------------------------------------------------------------
__global__ __launch_bounds__(256) void k_ln_residual(
    const float* __restrict__ x, const float* __restrict__ res,
    const float* __restrict__ g, const float* __restrict__ be,
    float* __restrict__ out, f16* __restrict__ outh, int Lc)
{
  long long row = blockIdx.x;
  const float* xr = x + row * (long long)Lc;
  const float* rr = res + row * (long long)Lc;
  __shared__ float sred[8];
  int tid = threadIdx.x, lane = tid & 31, wv = tid >> 5;

  float s = 0.0f;
  for (int c = tid; c < Lc; c += 256) s += xr[c] + rr[c];
  for (int off = 16; off >= 1; off >>= 1) s += __shfl_xor(s, off, 32);
  if (lane == 0) sred[wv] = s;
  __syncthreads();
  float mean = 0.0f;
  for (int i = 0; i < 8; ++i) mean += sred[i];
  mean /= (float)Lc;
  __syncthreads();

  float vs = 0.0f;
  for (int c = tid; c < Lc; c += 256) { float d = xr[c] + rr[c] - mean; vs += d * d; }
  for (int off = 16; off >= 1; off >>= 1) vs += __shfl_xor(vs, off, 32);
  if (lane == 0) sred[wv] = vs;
  __syncthreads();
  float var = 0.0f;
  for (int i = 0; i < 8; ++i) var += sred[i];
  float rstd = rsqrtf(var / (float)Lc + 1e-5f);

  for (int c = tid; c < Lc; c += 256) {
    float v = (xr[c] + rr[c] - mean) * rstd * g[c] + be[c];
    long long o = row * (long long)Lc + c;
    out[o] = v;
    if (outh) outh[o] = (f16)v;
  }
}

// ---------------------------------------------------------------------------
// Batched transpose with optional fused scalar combine:
//   out[b][c][r] = (f16)(in[b][r][c] * sum(wsum[0..7]) + badd[0])
// in: [Batch, R, C] fp32.  block (32,8), grid (C/32, R/32, Batch).
// ---------------------------------------------------------------------------
__global__ __launch_bounds__(256) void k_transpose_scale(
    const float* __restrict__ in, f16* __restrict__ out,
    int R, int C, const float* __restrict__ wsum, const float* __restrict__ badd)
{
  __shared__ float t[32][33];
  const int bz = blockIdx.z;
  in  += (long long)bz * R * C;
  out += (long long)bz * R * C;
  int c0 = blockIdx.x * 32, r0 = blockIdx.y * 32;
  for (int i = 0; i < 4; ++i) {
    int r = r0 + threadIdx.y + i * 8;
    t[threadIdx.y + i * 8][threadIdx.x] = in[(long long)r * C + c0 + threadIdx.x];
  }
  float alpha = 1.0f, beta = 0.0f;
  if (wsum) { float sw = 0.0f; for (int h = 0; h < 8; ++h) sw += wsum[h]; alpha = sw; beta = badd[0]; }
  __syncthreads();
  for (int i = 0; i < 4; ++i) {
    int c = c0 + threadIdx.y + i * 8;
    out[(long long)c * R + r0 + threadIdx.x] = (f16)(t[threadIdx.x][threadIdx.y + i * 8] * alpha + beta);
  }
}

__global__ void k_f32_to_f16(const float* __restrict__ in, f16* __restrict__ out, long long n)
{
  long long i = (long long)blockIdx.x * blockDim.x + threadIdx.x;
  long long stride = (long long)gridDim.x * blockDim.x;
  for (; i < n; i += stride) out[i] = (f16)in[i];
}

// Collapse Conv2d(1,H) weight [L,L,1,H] -> f16 [L,L] by summing over heads.
__global__ void k_sum_heads(const float* __restrict__ in, f16* __restrict__ out, int n)
{
  int i = blockIdx.x * blockDim.x + threadIdx.x;
  if (i < n) {
    float s = 0.0f;
    for (int h = 0; h < 8; ++h) s += in[(long long)i * 8 + h];
    out[i] = (f16)s;
  }
}

// ---------------------------------------------------------------------------
extern "C" void kernel_launch(void* const* d_in, const int* in_sizes, int n_in,
                              void* d_out, int out_size, void* d_ws, size_t ws_size,
                              hipStream_t stream)
{
  (void)in_sizes; (void)n_in; (void)out_size; (void)ws_size;

  constexpr int Bb = 32, Nd = 1024, Ld = 512, Hd = 8;
  const long long LL2 = (long long)Ld * Ld;       // 262144
  const long long NN2 = (long long)Nd * Nd;       // 1048576
  const long long sNL = (long long)Nd * Ld;       // per-batch activation stride
  const long long BNL = (long long)Bb * sNL;

  const float* x    = (const float*)d_in[0];
  const float* tWq  = (const float*)d_in[1];  const float* tbq = (const float*)d_in[2];
  const float* tWk  = (const float*)d_in[3];  const float* tbk = (const float*)d_in[4];
  const float* tWv  = (const float*)d_in[5];  const float* tbv = (const float*)d_in[6];
  const float* tWo  = (const float*)d_in[7];  const float* tbo = (const float*)d_in[8];
  const float* tg   = (const float*)d_in[9];  const float* tbt = (const float*)d_in[10];
  const float* cWq  = (const float*)d_in[11]; const float* cbq = (const float*)d_in[12];
  const float* cWk  = (const float*)d_in[13]; const float* cbk = (const float*)d_in[14];
  const float* cWv  = (const float*)d_in[15]; const float* cbv = (const float*)d_in[16];
  const float* cWo  = (const float*)d_in[17]; const float* cbo = (const float*)d_in[18];
  const float* cg   = (const float*)d_in[19]; const float* cbt = (const float*)d_in[20];
  const float* sWq  = (const float*)d_in[21]; const float* sbq = (const float*)d_in[22];
  const float* sWk  = (const float*)d_in[23]; const float* sbk = (const float*)d_in[24];
  const float* sWv  = (const float*)d_in[25]; const float* sbv = (const float*)d_in[26];
  const float* sW1  = (const float*)d_in[27]; const float* sb1 = (const float*)d_in[28];

  // ---- workspace layout --------------------------------------------------
  char* base = (char*)d_ws;
  size_t off = 0;
  auto alloc = [&](size_t bytes) -> void* {
    off = (off + 255) & ~(size_t)255;
    void* p = base + off; off += bytes; return p;
  };
  f16* w_tWq = (f16*)alloc(LL2 * 2);  f16* w_tWk = (f16*)alloc(LL2 * 2);
  f16* w_tWv = (f16*)alloc(LL2 * 2);  f16* w_cWq = (f16*)alloc(LL2 * 2);
  f16* w_cWk = (f16*)alloc(LL2 * 2);  f16* w_cWv = (f16*)alloc(LL2 * 2);
  f16* w_sWq = (f16*)alloc(NN2 * 2);  f16* w_sWk = (f16*)alloc(NN2 * 2);
  f16* w_sWv = (f16*)alloc(NN2 * 2);
  f16* w_WsT = (f16*)alloc(LL2 * 2);  f16* w_WsC = (f16*)alloc(LL2 * 2);
  f16* xh     = (f16*)alloc(BNL * 2);
  f16* xth    = (f16*)alloc(BNL * 2);
  f16* qh     = (f16*)alloc(BNL * 2);
  f16* kh     = (f16*)alloc(BNL * 2);
  f16* vh     = (f16*)alloc(BNL * 2);
  f16* lineH  = (f16*)alloc(BNL * 2);
  f16* toutH  = (f16*)alloc(BNL * 2);
  f16* slineH = (f16*)alloc(BNL * 2);
  f16*   attnH = (f16*)  alloc((long long)Bb * NN2 * 2);
  float* attnF = (float*)alloc((long long)Bb * NN2 * 4);
  float* lineF = (float*)alloc(BNL * 4);
  float* resF  = (float*)alloc(BNL * 4);
  float* toutF = (float*)alloc(BNL * 4);

  // ---- helpers -----------------------------------------------------------
  auto cvt = [&](const float* s, f16* d, long long n) {
    long long b = (n + 255) / 256; if (b > 8192) b = 8192;
    k_f32_to_f16<<<(int)b, 256, 0, stream>>>(s, d, n);
  };
  auto gemm = [&](const f16* A, int lda, long long sA,
                  const f16* Bm, int ldb, long long sB, int tB,
                  const float* bias, float alpha,
                  float* C, f16* Ch, int M, int N, int K) {
    dim3 g(N / GTN, M / GTM, Bb);
    if (tB)
      k_gemm_wmma<1><<<g, 256, 0, stream>>>(A, lda, sA, Bm, ldb, sB, bias, alpha,
                                            C, Ch, (long long)M * N, M, N, K);
    else
      k_gemm_wmma<0><<<g, 256, 0, stream>>>(A, lda, sA, Bm, ldb, sB, bias, alpha,
                                            C, Ch, (long long)M * N, M, N, K);
  };
  const float akd_t = 1.0f / sqrtf((float)Nd / Hd);   // 1/sqrt(128)
  const float akd_s = 1.0f / sqrtf((float)Ld / Hd);   // 1/8

  // ---- precompute: f16 weights, head-summed Wo, x / x^T in f16 ----------
  cvt(tWq, w_tWq, LL2); cvt(tWk, w_tWk, LL2); cvt(tWv, w_tWv, LL2);
  cvt(cWq, w_cWq, LL2); cvt(cWk, w_cWk, LL2); cvt(cWv, w_cWv, LL2);
  cvt(sWq, w_sWq, NN2); cvt(sWk, w_sWk, NN2); cvt(sWv, w_sWv, NN2);
  k_sum_heads<<<(int)(LL2 / 256), 256, 0, stream>>>(tWo, w_WsT, (int)LL2);
  k_sum_heads<<<(int)(LL2 / 256), 256, 0, stream>>>(cWo, w_WsC, (int)LL2);
  cvt(x, xh, BNL);
  { dim3 g(Ld / 32, Nd / 32, Bb);
    k_transpose_scale<<<g, dim3(32, 8), 0, stream>>>(x, xth, Nd, Ld, nullptr, nullptr); }

  // ---- Time_de_att: q/k/v = x @ W^T + b; attn = softmax(q k^T / kd) -----
  gemm(xh, Ld, sNL, w_tWq, Ld, 0, 1, tbq, 1.0f, nullptr, qh, Nd, Ld, Ld);
  gemm(xh, Ld, sNL, w_tWk, Ld, 0, 1, tbk, 1.0f, nullptr, kh, Nd, Ld, Ld);
  gemm(xh, Ld, sNL, w_tWv, Ld, 0, 1, tbv, 1.0f, nullptr, vh, Nd, Ld, Ld);
  gemm(qh, Ld, sNL, kh, Ld, sNL, 1, nullptr, akd_t, attnF, nullptr, Nd, Nd, Ld);
  k_softmax_rows<<<Bb * Nd, 256, 0, stream>>>(attnF, attnH, Nd);
  gemm(attnH, Nd, NN2, vh, Ld, sNL, 0, nullptr, 1.0f, nullptr, lineH, Nd, Ld, Nd);
  gemm(lineH, Ld, sNL, w_WsT, Ld, 0, 1, tbo, 1.0f, resF, nullptr, Nd, Ld, Ld);
  k_ln_residual<<<Bb * Nd, 256, 0, stream>>>(x, resF, tg, tbt, toutF, toutH, Ld);

  // ---- space_att2: over time axis, node features ------------------------
  gemm(xth, Nd, sNL, w_sWq, Nd, 0, 1, sbq, 1.0f, nullptr, qh, Ld, Nd, Nd);
  gemm(xth, Nd, sNL, w_sWk, Nd, 0, 1, sbk, 1.0f, nullptr, kh, Ld, Nd, Nd);
  gemm(xth, Nd, sNL, w_sWv, Nd, 0, 1, sbv, 1.0f, nullptr, vh, Ld, Nd, Nd);
  gemm(qh, Nd, sNL, kh, Nd, sNL, 1, nullptr, akd_s, attnF, nullptr, Ld, Ld, Nd);
  k_softmax_rows<<<Bb * Ld, 256, 0, stream>>>(attnF, attnH, Ld);
  gemm(attnH, Ld, LL2, vh, Nd, sNL, 0, nullptr, 1.0f, lineF, nullptr, Ld, Nd, Ld);
  { dim3 g(Nd / 32, Ld / 32, Bb);   // [B,L,N] -> [B,N,L], fused *sum(s_W1)+s_b1
    k_transpose_scale<<<g, dim3(32, 8), 0, stream>>>(lineF, slineH, Ld, Nd, sW1, sb1); }

  // ---- cross_de_att: q from space_out, k/v + residual from time_out -----
  gemm(slineH, Ld, sNL, w_cWq, Ld, 0, 1, cbq, 1.0f, nullptr, qh, Nd, Ld, Ld);
  gemm(toutH,  Ld, sNL, w_cWk, Ld, 0, 1, cbk, 1.0f, nullptr, kh, Nd, Ld, Ld);
  gemm(toutH,  Ld, sNL, w_cWv, Ld, 0, 1, cbv, 1.0f, nullptr, vh, Nd, Ld, Ld);
  gemm(qh, Ld, sNL, kh, Ld, sNL, 1, nullptr, akd_t, attnF, nullptr, Nd, Nd, Ld);
  k_softmax_rows<<<Bb * Nd, 256, 0, stream>>>(attnF, attnH, Nd);
  gemm(attnH, Nd, NN2, vh, Ld, sNL, 0, nullptr, 1.0f, nullptr, lineH, Nd, Ld, Nd);
  gemm(lineH, Ld, sNL, w_WsC, Ld, 0, 1, cbo, 1.0f, resF, nullptr, Nd, Ld, Ld);
  k_ln_residual<<<Bb * Nd, 256, 0, stream>>>(toutF, resF, cg, cbt,
                                             (float*)d_out, nullptr, Ld);
}
